// MultiHeadSelfAttention_11897059410172
// MI455X (gfx1250) — matrix operand order repaired and checked by reading, not compiled, not run on hardware
//
#include <hip/hip_runtime.h>
#include <hip/hip_bf16.h>

typedef __bf16 bf16;
typedef __attribute__((ext_vector_type(16))) __bf16 v16bf;
typedef __attribute__((ext_vector_type(8)))  __bf16 v8bf;
typedef __attribute__((ext_vector_type(2)))  __bf16 v2bf;
typedef __attribute__((ext_vector_type(8)))  float  v8f;
typedef __attribute__((ext_vector_type(4)))  unsigned int u32x4;
typedef __attribute__((ext_vector_type(8)))  int i32x8;
typedef __attribute__((ext_vector_type(4)))  int i32x4;

#define B_  4
#define T_  2048
#define D_  1024
#define H_  16
#define DH_ 64
#define N3_ 3072

static __device__ __forceinline__ v8f zero8() {
    v8f z;
#pragma unroll
    for (int i = 0; i < 8; ++i) z[i] = 0.0f;
    return z;
}

static __device__ __forceinline__ v16bf mk16(v8bf lo, v8bf hi) {
    return __builtin_shufflevector(lo, hi, 0,1,2,3,4,5,6,7,8,9,10,11,12,13,14,15);
}

// ---------------------------------------------------------------------------
// TDM 2D tile load (global bf16 -> LDS), with LDS padding done by the DMA.
// D# per CDNA5 ISA ch.8: group0 = {count, lds_addr, global_addr, type=2},
// group1 = {data_size=2B, pad_enable, pad codes, tensor/tile dims, stride}.
// This toolchain's builtin is the 6-arg form:
//   (u32x4 g0, i32x8 g1, i32x4 g2, i32x4 g3, i32x8 extra, i32 cpol)
// ---------------------------------------------------------------------------
static __device__ __forceinline__ void tdm_load_2d(
    unsigned lds_addr, const void* gptr,
    unsigned tensor_d0, unsigned tensor_d1,
    unsigned tile_d0, unsigned tile_d1,
    unsigned stride0, unsigned pad_int_code, unsigned pad_amt_code)
{
    unsigned long long ga = (unsigned long long)(uintptr_t)gptr;
    u32x4 g0;
    g0[0] = 1u;                                   // count = 1 valid descriptor
    g0[1] = lds_addr;                             // LDS byte address
    g0[2] = (unsigned)ga;                         // global_addr[31:0]
    g0[3] = (unsigned)((ga >> 32) & 0x01FFFFFFu)  // global_addr[56:32]
            | (2u << 30);                         // type = 2 ("image")
    i32x8 g1;
    g1[0] = (int)((1u << 16)                      // data_size = 2 bytes
                  | (1u << 20)                    // pad_enable
                  | (pad_int_code << 22)          // pad_interval
                  | (pad_amt_code << 25));        // pad_amount
    g1[1] = (int)(tensor_d0 << 16);                        // tensor_dim0[15:0]
    g1[2] = (int)((tensor_d0 >> 16) | (tensor_d1 << 16));  // dim0 hi | dim1 lo
    g1[3] = (int)((tensor_d1 >> 16) | (tile_d0 << 16));    // dim1 hi | tile0
    g1[4] = (int)(tile_d1 & 0xFFFFu);                      // tile1 | tile2=0
    g1[5] = (int)stride0;                                  // dim0_stride lo
    g1[6] = 0;
    g1[7] = 0;
    i32x4 z4; z4[0] = 0; z4[1] = 0; z4[2] = 0; z4[3] = 0;
    i32x8 z8;
#pragma unroll
    for (int i = 0; i < 8; ++i) z8[i] = 0;
    __builtin_amdgcn_tensor_load_to_lds(g0, g1, z4, z4, z8, 0);
}

// ---------------------------------------------------------------------------
// Kernel 1: qkv = x @ Wqkv + bqkv  (fp32 in, bf16 Q/K/V out)
// Q,K head-major [bh][t][dh]; V stored TRANSPOSED [bh][dh][t] so the
// attention kernel can TDM-load V tiles contiguously.
// ---------------------------------------------------------------------------
__global__ __launch_bounds__(256) void qkv_gemm(
    const float* __restrict__ x, const float* __restrict__ Wqkv,
    const float* __restrict__ bqkv,
    bf16* __restrict__ Qb, bf16* __restrict__ Kb, bf16* __restrict__ Vb)
{
    __shared__ __align__(16) bf16 As[64][40];
    __shared__ __align__(16) bf16 Bs[128][40];

    const int tid  = threadIdx.x;
    const int lane = tid & 31;
    const int wave = tid >> 5;
    const int half = lane >> 4;
    const int mr   = lane & 15;
    const int wm   = wave & 3;
    const int wn   = wave >> 2;
    const int m0   = blockIdx.y * 64;
    const int n0   = blockIdx.x * 128;

    v8f acc[4];
#pragma unroll
    for (int i = 0; i < 4; ++i) acc[i] = zero8();

    const int am  = tid >> 2, aseg = tid & 3;
    const int bk2 = tid & 15, bn8 = tid >> 4;

    for (int k0 = 0; k0 < D_; k0 += 32) {
        {
            const float* src = x + (size_t)(m0 + am) * D_ + k0 + aseg * 8;
            if (k0 + 32 < D_) __builtin_prefetch(src + 32, 0, 1);
            v8bf t;
#pragma unroll
            for (int i = 0; i < 8; ++i) t[i] = (bf16)src[i];
            *(v8bf*)&As[am][aseg * 8] = t;
        }
        {
            const float* r0 = Wqkv + (size_t)(k0 + bk2 * 2) * N3_ + n0 + bn8 * 8;
            const float* r1 = r0 + N3_;
#pragma unroll
            for (int i = 0; i < 8; ++i) {
                v2bf p; p[0] = (bf16)r0[i]; p[1] = (bf16)r1[i];
                *(v2bf*)&Bs[bn8 * 8 + i][bk2 * 2] = p;
            }
        }
        __syncthreads();

        v16bf a = mk16(*(const v8bf*)&As[wm * 16 + mr][half * 8],
                       *(const v8bf*)&As[wm * 16 + mr][16 + half * 8]);
#pragma unroll
        for (int nt = 0; nt < 4; ++nt) {
            const int n = wn * 64 + nt * 16 + mr;
            v16bf b = mk16(*(const v8bf*)&Bs[n][half * 8],
                           *(const v8bf*)&Bs[n][16 + half * 8]);
            acc[nt] = __builtin_amdgcn_wmma_f32_16x16x32_bf16(
                false, a, false, b, (short)0, acc[nt], false, false);
        }
        __syncthreads();
    }

#pragma unroll
    for (int nt = 0; nt < 4; ++nt) {
        const int n_g   = n0 + wn * 64 + nt * 16 + mr;
        const int which = n_g >> 10;                  // 0=Q 1=K 2=V
        const int c     = n_g & 1023;
        const int h     = c >> 6, dh = c & 63;
        const float bias = bqkv[n_g];
        bf16* dst = (which == 0) ? Qb : ((which == 1) ? Kb : Vb);
#pragma unroll
        for (int r = 0; r < 8; ++r) {
            const int m_g = m0 + wm * 16 + half * 8 + r;
            const int b   = m_g >> 11, t = m_g & (T_ - 1);
            const size_t idx = (which == 2)
                ? ((size_t)(b * H_ + h) * DH_ + dh) * T_ + t       // V: [dh][t]
                : ((size_t)(b * H_ + h) * T_ + t) * DH_ + dh;      // Q/K: [t][dh]
            dst[idx] = (bf16)(acc[nt][r] + bias);
        }
    }
}

// ---------------------------------------------------------------------------
// Kernel 2: flash attention. K/V tiles are DMA'd into LDS by the Tensor Data
// Mover (double-buffered, TENSORcnt-synchronized); TDM pad_enable reproduces
// the bank-conflict padding (K stride 144B, V stride 80B).
// ---------------------------------------------------------------------------
__global__ __launch_bounds__(256) void attn_kernel(
    const bf16* __restrict__ Qb, const bf16* __restrict__ Kb,
    const bf16* __restrict__ Vb, const unsigned char* __restrict__ pad,
    bf16* __restrict__ Ob)
{
    __shared__ __align__(16) bf16 Ks[2][32][72];     // [key][dh], 144B rows
    __shared__ __align__(16) bf16 Vs[2][64][40];     // [dh][key],  80B rows
    __shared__ __align__(16) bf16 Ps[8][16][40];     // per-wave P relayout

    const int tid  = threadIdx.x;
    const int lane = tid & 31;
    const int wave = tid >> 5;
    const int half = lane >> 4;
    const int mr   = lane & 15;
    const int bh   = blockIdx.y;
    const int b    = bh >> 4, h = bh & 15;
    const int qb   = blockIdx.x;
    const int q0   = qb * 128 + wave * 16;
    const size_t headoff = (size_t)bh * T_ * DH_;

    // Q fragments (two k-steps over dh=64), pre-scaled by 1/sqrt(DH)=0.125
    v16bf aq[2];
#pragma unroll
    for (int j = 0; j < 2; ++j) {
        const bf16* ql = Qb + headoff + (size_t)(q0 + mr) * DH_ + j * 32 + half * 8;
        v8bf lo = *(const v8bf*)ql;
        v8bf hi = *(const v8bf*)(ql + 16);
#pragma unroll
        for (int i = 0; i < 8; ++i) {
            aq[j][i]     = (bf16)((float)lo[i] * 0.125f);
            aq[j][8 + i] = (bf16)((float)hi[i] * 0.125f);
        }
    }

    v8f oacc[4];
#pragma unroll
    for (int i = 0; i < 4; ++i) oacc[i] = zero8();
    float runmax[8], runsum[8];
#pragma unroll
    for (int r = 0; r < 8; ++r) { runmax[r] = -__builtin_inff(); runsum[r] = 0.0f; }

    const unsigned ksLds[2] = { (unsigned)(uintptr_t)&Ks[0][0][0],
                                (unsigned)(uintptr_t)&Ks[1][0][0] };
    const unsigned vsLds[2] = { (unsigned)(uintptr_t)&Vs[0][0][0],
                                (unsigned)(uintptr_t)&Vs[1][0][0] };

    const int ntiles = ((qb + 1) * 128) / 32;

    // prologue: DMA tile 0
    if (wave == 0) {
        // K tile: 32 rows x 64 dh from [t][dh]; pad 16B per 128B row -> 144B
        tdm_load_2d(ksLds[0], Kb + headoff, 64, T_, 64, 32, 64, 4, 3);
        // V tile: 64 rows x 32 keys from [dh][t]; pad 16B per 64B row -> 80B
        tdm_load_2d(vsLds[0], Vb + headoff, T_, 64, 32, 64, T_, 3, 3);
    }

    for (int it = 0; it < ntiles; ++it) {
        const int k0   = it * 32;
        const int bufc = it & 1;
        if (wave == 0) {
            if (it + 1 < ntiles) {
                const int kn = (it + 1) * 32;
                const int bn = (it + 1) & 1;
                tdm_load_2d(ksLds[bn], Kb + headoff + (size_t)kn * DH_,
                            64, T_, 64, 32, 64, 4, 3);
                tdm_load_2d(vsLds[bn], Vb + headoff + (size_t)kn,
                            T_, 64, 32, 64, T_, 3, 3);
                __builtin_amdgcn_s_wait_tensorcnt(2);  // tile `it` landed
            } else {
                __builtin_amdgcn_s_wait_tensorcnt(0);
            }
        }
        __syncthreads();

        if (k0 <= q0 + 15) {                       // wave-uniform causal skip
            // ---- S = Q K^T (16 queries x 32 keys) ----
            v8f sacc[2];
            sacc[0] = zero8(); sacc[1] = zero8();
#pragma unroll
            for (int nt = 0; nt < 2; ++nt) {
#pragma unroll
                for (int j = 0; j < 2; ++j) {
                    v16bf bk = mk16(
                        *(const v8bf*)&Ks[bufc][nt * 16 + mr][j * 32 + half * 8],
                        *(const v8bf*)&Ks[bufc][nt * 16 + mr][j * 32 + 16 + half * 8]);
                    sacc[nt] = __builtin_amdgcn_wmma_f32_16x16x32_bf16(
                        false, aq[j], false, bk, (short)0, sacc[nt], false, false);
                }
            }
            // ---- mask + online softmax (16-lane row reductions) ----
            const int kg0 = k0 + mr, kg1 = k0 + 16 + mr;
            const bool pm0 = pad[b * T_ + kg0] != 0;
            const bool pm1 = pad[b * T_ + kg1] != 0;
            float scale[8], p0s[8], p1s[8];
#pragma unroll
            for (int r = 0; r < 8; ++r) {
                const int qg = q0 + half * 8 + r;
                const bool ok0 = pm0 && (kg0 <= qg);
                const bool ok1 = pm1 && (kg1 <= qg);
                float t = fmaxf(ok0 ? sacc[0][r] : -__builtin_inff(),
                                ok1 ? sacc[1][r] : -__builtin_inff());
#pragma unroll
                for (int off = 1; off < 16; off <<= 1)
                    t = fmaxf(t, __shfl_xor(t, off, 32));
                const float nm = fmaxf(runmax[r], t);
                const float sc = (runmax[r] == -__builtin_inff())
                                     ? 0.0f : __expf(runmax[r] - nm);
                const float p0 = ok0 ? __expf(sacc[0][r] - nm) : 0.0f;
                const float p1 = ok1 ? __expf(sacc[1][r] - nm) : 0.0f;
                float rs = p0 + p1;
#pragma unroll
                for (int off = 1; off < 16; off <<= 1)
                    rs += __shfl_xor(rs, off, 32);
                runsum[r] = runsum[r] * sc + rs;
                runmax[r] = nm;
                scale[r]  = sc;
                p0s[r] = p0; p1s[r] = p1;
            }
            // ---- relayout P (C-frag -> A-frag) through per-wave LDS ----
#pragma unroll
            for (int r = 0; r < 8; ++r) {
                Ps[wave][half * 8 + r][mr]      = (bf16)p0s[r];
                Ps[wave][half * 8 + r][16 + mr] = (bf16)p1s[r];
            }
            asm volatile("" ::: "memory");   // DS in-order within a wave
            v16bf pa = mk16(*(const v8bf*)&Ps[wave][mr][half * 8],
                            *(const v8bf*)&Ps[wave][mr][16 + half * 8]);
            // ---- O = O*scale + P V ----
#pragma unroll
            for (int dt = 0; dt < 4; ++dt) {
#pragma unroll
                for (int r = 0; r < 8; ++r) oacc[dt][r] *= scale[r];
                v16bf vb = mk16(
                    *(const v8bf*)&Vs[bufc][dt * 16 + mr][half * 8],
                    *(const v8bf*)&Vs[bufc][dt * 16 + mr][16 + half * 8]);
                oacc[dt] = __builtin_amdgcn_wmma_f32_16x16x32_bf16(
                    false, pa, false, vb, (short)0, oacc[dt], false, false);
            }
        }
        __syncthreads();
    }

    // ---- normalize + store bf16 attn-out in (B,T,D) layout ----
#pragma unroll
    for (int dt = 0; dt < 4; ++dt) {
#pragma unroll
        for (int r = 0; r < 8; ++r) {
            const int qg = q0 + half * 8 + r;
            const float inv = (runsum[r] > 0.0f) ? (1.0f / runsum[r]) : 0.0f;
            Ob[((size_t)(b * T_ + qg)) * D_ + h * DH_ + dt * 16 + mr] =
                (bf16)(oacc[dt][r] * inv);
        }
    }
}

// ---------------------------------------------------------------------------
// Kernel 3: out = attn_out @ Wout + bout
// ---------------------------------------------------------------------------
__global__ __launch_bounds__(256) void out_gemm(
    const bf16* __restrict__ Ob, const float* __restrict__ Wout,
    const float* __restrict__ bout, float* __restrict__ out)
{
    __shared__ __align__(16) bf16 As[64][40];
    __shared__ __align__(16) bf16 Bs[128][40];

    const int tid  = threadIdx.x;
    const int lane = tid & 31;
    const int wave = tid >> 5;
    const int half = lane >> 4;
    const int mr   = lane & 15;
    const int wm   = wave & 3;
    const int wn   = wave >> 2;
    const int m0   = blockIdx.y * 64;
    const int n0   = blockIdx.x * 128;

    v8f acc[4];
#pragma unroll
    for (int i = 0; i < 4; ++i) acc[i] = zero8();

    const int am  = tid >> 2, aseg = tid & 3;
    const int bk2 = tid & 15, bn8 = tid >> 4;

    for (int k0 = 0; k0 < D_; k0 += 32) {
        *(v8bf*)&As[am][aseg * 8] =
            *(const v8bf*)(Ob + (size_t)(m0 + am) * D_ + k0 + aseg * 8);
        {
            const float* r0 = Wout + (size_t)(k0 + bk2 * 2) * D_ + n0 + bn8 * 8;
            const float* r1 = r0 + D_;
            if (k0 + 32 < D_) __builtin_prefetch(r0 + 32 * D_, 0, 1);
#pragma unroll
            for (int i = 0; i < 8; ++i) {
                v2bf p; p[0] = (bf16)r0[i]; p[1] = (bf16)r1[i];
                *(v2bf*)&Bs[bn8 * 8 + i][bk2 * 2] = p;
            }
        }
        __syncthreads();

        v16bf a = mk16(*(const v8bf*)&As[wm * 16 + mr][half * 8],
                       *(const v8bf*)&As[wm * 16 + mr][16 + half * 8]);
#pragma unroll
        for (int nt = 0; nt < 4; ++nt) {
            const int n = wn * 64 + nt * 16 + mr;
            v16bf b = mk16(*(const v8bf*)&Bs[n][half * 8],
                           *(const v8bf*)&Bs[n][16 + half * 8]);
            acc[nt] = __builtin_amdgcn_wmma_f32_16x16x32_bf16(
                false, a, false, b, (short)0, acc[nt], false, false);
        }
        __syncthreads();
    }

#pragma unroll
    for (int nt = 0; nt < 4; ++nt) {
        const int n_g = n0 + wn * 64 + nt * 16 + mr;
        const float bias = bout[n_g];
#pragma unroll
        for (int r = 0; r < 8; ++r) {
            const int m_g = m0 + wm * 16 + half * 8 + r;
            out[(size_t)m_g * D_ + n_g] = acc[nt][r] + bias;
        }
    }
}

// ---------------------------------------------------------------------------
extern "C" void kernel_launch(void* const* d_in, const int* in_sizes, int n_in,
                              void* d_out, int out_size, void* d_ws, size_t ws_size,
                              hipStream_t stream)
{
    const float*         x    = (const float*)d_in[0];
    const unsigned char* pad  = (const unsigned char*)d_in[1];
    const float*         Wqkv = (const float*)d_in[2];
    const float*         bqkv = (const float*)d_in[3];
    const float*         Wout = (const float*)d_in[4];
    const float*         bout = (const float*)d_in[5];
    float*               out  = (float*)d_out;

    char* ws = (char*)d_ws;
    const size_t headBytes = (size_t)B_ * H_ * T_ * DH_ * sizeof(bf16);  // 16 MB
    bf16* Qb = (bf16*)(ws);
    bf16* Kb = (bf16*)(ws + headBytes);
    bf16* Vb = (bf16*)(ws + 2 * headBytes);
    bf16* Ob = (bf16*)(ws + 3 * headBytes);

    qkv_gemm<<<dim3(N3_ / 128, (B_ * T_) / 64), 256, 0, stream>>>(
        x, Wqkv, bqkv, Qb, Kb, Vb);
    attn_kernel<<<dim3(T_ / 128, B_ * H_), 256, 0, stream>>>(
        Qb, Kb, Vb, pad, Ob);
    out_gemm<<<dim3(D_ / 128, (B_ * T_) / 64), 256, 0, stream>>>(
        Ob, Wout, bout, out);
}
